// TransformerLayer_41489384079528
// MI455X (gfx1250) — compile-verified, matrix operand order checked by reading, not run on hardware
//
#include <hip/hip_runtime.h>

typedef __attribute__((ext_vector_type(16))) _Float16 v16h;
typedef __attribute__((ext_vector_type(8)))  float    v8f;

static constexpr int B_ = 32, S_ = 512, E_ = 512, H_ = 8, D_ = 64;
static constexpr int M_ = B_ * S_;   // 16384 rows

union F16Frag { v16h h; uint4 u[2]; };

// Load a 16x32 f16 fragment in the CDNA5 WMMA A/B register layout.
// Matrix is row-major with leading dim `ld`; for the A-matrix row = M index,
// for the B-matrix (stored as W, i.e. row = N index, contiguous in K) the
// same layout yields the required column-major K x 16 operand.
// Lane l (0..31): row = row0 + (l&15); K elems [c..c+7] and [c+16..c+23],
// where c = k0 + (l>=16 ? 8 : 0).
__device__ __forceinline__ v16h ldfrag(const _Float16* __restrict__ p, int ld,
                                       int row0, int k0, int lane) {
  const int r = row0 + (lane & 15);
  const int c = k0 + ((lane >> 4) << 3);
  const _Float16* q = p + (size_t)r * ld + c;
  F16Frag f;
  f.u[0] = *reinterpret_cast<const uint4*>(q);
  f.u[1] = *reinterpret_cast<const uint4*>(q + 16);
  return f.h;
}

__device__ __forceinline__ v8f wmma_f16(v16h a, v16h b, v8f c) {
  // D(f32 16x16) = A(f16 16x32) * B(f16 32x16) + C
  return __builtin_amdgcn_wmma_f32_16x16x32_f16(false, a, false, b,
                                                (short)0, c, false, false);
}

// ---------------------------------------------------------------- conversions
__global__ void cvt_f16(const float* __restrict__ in, _Float16* __restrict__ out,
                        int n) {
  int i = blockIdx.x * blockDim.x + threadIdx.x;
  const int stride = gridDim.x * blockDim.x;
  for (; i < n; i += stride) out[i] = (_Float16)in[i];
}

// ------------------------------------------------------------------- GEMM
// C[M,N] = A[M,K] * W[N,K]^T  (+bias, relu, +resid) -> f32 and/or f16 out.
// MODE bits: 1=bias  2=relu  4=add f32 residual  8=write f32  16=write f16
// Block = 4 waves (2x2), each wave owns a 64x64 tile (4x4 WMMA tiles):
// 16 WMMAs per 8KB of fragment loads = 32 FLOP/byte, 4x fragment reuse.
template <int MODE>
__global__ __launch_bounds__(128)
void gemm_wmma(const _Float16* __restrict__ A, const _Float16* __restrict__ W,
               const float* __restrict__ bias, const float* __restrict__ resid,
               float* __restrict__ Cf, _Float16* __restrict__ Ch,
               int M, int N, int K) {
  const int lane = threadIdx.x & 31;
  const int wave = threadIdx.x >> 5;                 // 4 waves: 2(M) x 2(N)
  const int m0 = blockIdx.y * 128 + (wave >> 1) * 64;
  const int n0 = blockIdx.x * 128 + (wave & 1) * 64;
  const int r    = lane & 15;
  const int coff = (lane >> 4) << 3;

  v8f z = {};
  v8f c[4][4];
#pragma unroll
  for (int i = 0; i < 4; ++i)
#pragma unroll
    for (int j = 0; j < 4; ++j) c[i][j] = z;

  for (int kk = 0; kk < K; kk += 32) {
    v16h a[4], b[4];
#pragma unroll
    for (int i = 0; i < 4; ++i) a[i] = ldfrag(A, K, m0 + 16 * i, kk, lane);
#pragma unroll
    for (int j = 0; j < 4; ++j) b[j] = ldfrag(W, K, n0 + 16 * j, kk, lane);

    if (kk + 64 < K) {  // prefetch 2 k-steps ahead -> global_prefetch_b8
      const int pc = kk + 64 + coff;
#pragma unroll
      for (int i = 0; i < 4; ++i)
        __builtin_prefetch(A + (size_t)(m0 + 16 * i + r) * K + pc, 0, 3);
#pragma unroll
      for (int j = 0; j < 4; ++j)
        __builtin_prefetch(W + (size_t)(n0 + 16 * j + r) * K + pc, 0, 3);
    }

#pragma unroll
    for (int i = 0; i < 4; ++i)
#pragma unroll
      for (int j = 0; j < 4; ++j)
        c[i][j] = wmma_f16(a[i], b[j], c[i][j]);
  }

  // C layout: lane holds n = lane&15; VGPR v holds m = v + 8*(lane>=16)
#pragma unroll
  for (int i = 0; i < 4; ++i) {
#pragma unroll
    for (int j = 0; j < 4; ++j) {
      const int nn = n0 + j * 16 + r;
      const int mb = m0 + i * 16 + coff;
      const float bv = (MODE & 1) ? bias[nn] : 0.0f;
#pragma unroll
      for (int v = 0; v < 8; ++v) {
        float t = c[i][j][v] + bv;
        if (MODE & 2) t = fmaxf(t, 0.0f);
        const size_t idx = (size_t)(mb + v) * N + nn;
        if (MODE & 4) t += resid[idx];
        if (MODE & 8) Cf[idx] = t;
        if (MODE & 16) Ch[idx] = (_Float16)t;
      }
    }
  }
}

// --------------------------------------------------------------- qkv repack
// qkv f16 [B*S, 3E] -> Q,K f16 [B*H, S, D], V transposed f16 [B*H, D, S]
__global__ void repack_qkv(const _Float16* __restrict__ qkv,
                           _Float16* __restrict__ Q, _Float16* __restrict__ K,
                           _Float16* __restrict__ Vt) {
  const int i = blockIdx.x * blockDim.x + threadIdx.x;   // over B*S*E
  const int e = i & (E_ - 1);
  const int s = (i >> 9) & (S_ - 1);
  const int b = i >> 18;
  const int h = e >> 6;
  const int d = e & (D_ - 1);
  const size_t src = (size_t)(b * S_ + s) * (3 * E_) + e;
  const size_t bh = (size_t)b * H_ + h;
  Q [(bh * S_ + s) * D_ + d] = qkv[src];
  K [(bh * S_ + s) * D_ + d] = qkv[src + E_];
  Vt[(bh * D_ + d) * S_ + s] = qkv[src + 2 * E_];
}

// --------------------------------------------------------------- attention
// One wave per (head, 16-query tile). Flash-style online softmax, causal.
__global__ __launch_bounds__(32)
void attn_wmma(const _Float16* __restrict__ Q, const _Float16* __restrict__ Km,
               const _Float16* __restrict__ Vt, _Float16* __restrict__ ctx) {
  const int lane = threadIdx.x;
  const int half = lane >> 4;         // 0/1
  const int l16  = lane & 15;
  const int q0 = blockIdx.x * 16;
  const int bh = blockIdx.y;
  const int b = bh >> 3, h = bh & 7;

  const _Float16* Qh = Q  + (size_t)bh * S_ * D_;   // [S,D]
  const _Float16* Kh = Km + (size_t)bh * S_ * D_;   // [S,D]
  const _Float16* Vh = Vt + (size_t)bh * D_ * S_;   // [D,S]

  // Q tile as two 16x32 A-fragments (K-dim = D = 64)
  const v16h aq0 = ldfrag(Qh, D_, q0, 0,  lane);
  const v16h aq1 = ldfrag(Qh, D_, q0, 32, lane);

  v8f z = {};
  v8f acc[4] = {z, z, z, z};          // ctx accum, 4 d-tiles of 16
  float rm[8], rs[8];
#pragma unroll
  for (int v = 0; v < 8; ++v) { rm[v] = -3.0e38f; rs[v] = 0.0f; }

  __shared__ __align__(16) _Float16 pbuf[16 * 32];  // P transpose staging
  const float scale = 0.125f;                        // 1/sqrt(D)

  for (int kk = 0; kk < q0 + 16; kk += 32) {
    // scores for two 16-key tiles
    v8f s0 = z, s1 = z;
    {
      v16h k0lo = ldfrag(Kh, D_, kk,      0,  lane);
      v16h k0hi = ldfrag(Kh, D_, kk,      32, lane);
      s0 = wmma_f16(aq0, k0lo, s0);
      s0 = wmma_f16(aq1, k0hi, s0);
      v16h k1lo = ldfrag(Kh, D_, kk + 16, 0,  lane);
      v16h k1hi = ldfrag(Kh, D_, kk + 16, 32, lane);
      s1 = wmma_f16(aq0, k1lo, s1);
      s1 = wmma_f16(aq1, k1hi, s1);
    }

    const int j0 = kk + l16, j1 = kk + 16 + l16;
    float fac[8];
#pragma unroll
    for (int v = 0; v < 8; ++v) {
      const int m = q0 + v + (half << 3);           // query row
      float e0 = (j0 <= m) ? s0[v] * scale : -3.0e38f;
      float e1 = (j1 <= m) ? s1[v] * scale : -3.0e38f;
      float mx = fmaxf(e0, e1);
#pragma unroll
      for (int off = 1; off < 16; off <<= 1)        // row max across half-wave
        mx = fmaxf(mx, __shfl_xor(mx, off, 32));
      const float nm = fmaxf(rm[v], mx);
      const float f  = __expf(rm[v] - nm);
      const float p0 = __expf(e0 - nm);
      const float p1 = __expf(e1 - nm);
      float ps = p0 + p1;
#pragma unroll
      for (int off = 1; off < 16; off <<= 1)        // row sum across half-wave
        ps += __shfl_xor(ps, off, 32);
      rs[v] = rs[v] * f + ps;
      rm[v] = nm;
      fac[v] = f;
      const int mr = v + (half << 3);
      pbuf[mr * 32 + l16]      = (_Float16)p0;      // C-layout -> [row][key]
      pbuf[mr * 32 + 16 + l16] = (_Float16)p1;
    }
#pragma unroll
    for (int dt = 0; dt < 4; ++dt)
#pragma unroll
      for (int v = 0; v < 8; ++v) acc[dt][v] *= fac[v];

    __syncthreads();
    // read P back in A-fragment layout (row = l16, K elems per half)
    F16Frag pf;
    const _Float16* pr = pbuf + l16 * 32 + (half << 3);
    pf.u[0] = *reinterpret_cast<const uint4*>(pr);
    pf.u[1] = *reinterpret_cast<const uint4*>(pr + 16);

#pragma unroll
    for (int dt = 0; dt < 4; ++dt) {
      v16h bv = ldfrag(Vh, S_, dt * 16, kk, lane);  // V^T rows = d, contig keys
      acc[dt] = wmma_f16(pf.h, bv, acc[dt]);
    }
    __syncthreads();
  }

  // normalize and scatter ctx back to [B*S, E] interleaved-head layout
#pragma unroll
  for (int v = 0; v < 8; ++v) {
    const float inv = 1.0f / rs[v];
    const int m = q0 + v + (half << 3);
#pragma unroll
    for (int dt = 0; dt < 4; ++dt) {
      const int d = dt * 16 + l16;
      ctx[(size_t)(b * S_ + m) * E_ + h * D_ + d] = (_Float16)(acc[dt][v] * inv);
    }
  }
}

// --------------------------------------------------------------- layernorm
__global__ __launch_bounds__(128)
void ln_rows(const float* __restrict__ in, const float* __restrict__ gam,
             const float* __restrict__ bet, float* __restrict__ outf,
             _Float16* __restrict__ outh) {
  const int row = blockIdx.x;
  const int tid = threadIdx.x;                       // 128 threads x 4 elems
  const float4 v = reinterpret_cast<const float4*>(in + (size_t)row * E_)[tid];
  float s  = v.x + v.y + v.z + v.w;
  float ss = v.x * v.x + v.y * v.y + v.z * v.z + v.w * v.w;
#pragma unroll
  for (int off = 16; off > 0; off >>= 1) {
    s  += __shfl_xor(s,  off, 32);
    ss += __shfl_xor(ss, off, 32);
  }
  __shared__ float red[8];
  if ((tid & 31) == 0) { red[tid >> 5] = s; red[4 + (tid >> 5)] = ss; }
  __syncthreads();
  s  = red[0] + red[1] + red[2] + red[3];
  ss = red[4] + red[5] + red[6] + red[7];
  const float mean = s * (1.0f / E_);
  const float inv  = rsqrtf(ss * (1.0f / E_) - mean * mean + 1e-5f);
  const int c = tid * 4;
  float4 ov;
  ov.x = (v.x - mean) * inv * gam[c + 0] + bet[c + 0];
  ov.y = (v.y - mean) * inv * gam[c + 1] + bet[c + 1];
  ov.z = (v.z - mean) * inv * gam[c + 2] + bet[c + 2];
  ov.w = (v.w - mean) * inv * gam[c + 3] + bet[c + 3];
  reinterpret_cast<float4*>(outf + (size_t)row * E_)[tid] = ov;
  if (outh) {
    outh[(size_t)row * E_ + c + 0] = (_Float16)ov.x;
    outh[(size_t)row * E_ + c + 1] = (_Float16)ov.y;
    outh[(size_t)row * E_ + c + 2] = (_Float16)ov.z;
    outh[(size_t)row * E_ + c + 3] = (_Float16)ov.w;
  }
}

// ------------------------------------------------------------------ launch
extern "C" void kernel_launch(void* const* d_in, const int* in_sizes, int n_in,
                              void* d_out, int out_size, void* d_ws, size_t ws_size,
                              hipStream_t stream) {
  const float* x    = (const float*)d_in[0];
  const float* Wqkv = (const float*)d_in[1];
  const float* bqkv = (const float*)d_in[2];
  const float* Wout = (const float*)d_in[3];
  const float* bout = (const float*)d_in[4];
  const float* W1   = (const float*)d_in[5];
  const float* b1   = (const float*)d_in[6];
  const float* W2   = (const float*)d_in[7];
  const float* b2   = (const float*)d_in[8];
  const float* g1   = (const float*)d_in[9];
  const float* be1  = (const float*)d_in[10];
  const float* g2   = (const float*)d_in[11];
  const float* be2  = (const float*)d_in[12];
  float* out = (float*)d_out;

  char* ws = (char*)d_ws;
  const size_t MB = 1ull << 20;
  _Float16* x_h   = (_Float16*)(ws + 0 * MB);     // 16 MB
  _Float16* Wqkvh = (_Float16*)(ws + 16 * MB);    // 1.5 MB
  _Float16* Wouth = (_Float16*)(ws + 18 * MB);    // 0.5 MB
  _Float16* W1h   = (_Float16*)(ws + 19 * MB);    // 2 MB
  _Float16* W2h   = (_Float16*)(ws + 21 * MB);    // 2 MB
  _Float16* qkv   = (_Float16*)(ws + 23 * MB);    // 48 MB
  _Float16* Qb    = (_Float16*)(ws + 71 * MB);    // 16 MB
  _Float16* Kb    = (_Float16*)(ws + 87 * MB);    // 16 MB
  _Float16* Vtb   = (_Float16*)(ws + 103 * MB);   // 16 MB
  _Float16* ctx   = (_Float16*)(ws + 119 * MB);   // 16 MB
  float*    y1    = (float*)   (ws + 135 * MB);   // 32 MB
  float*    x1f   = (float*)   (ws + 167 * MB);   // 32 MB
  _Float16* x1h   = (_Float16*)(ws + 199 * MB);   // 16 MB  (peak 215 MB)
  _Float16* h1    = (_Float16*)(ws + 23 * MB);    // 64 MB, reuses dead qkv+Q
  float*    y2    = (float*)   (ws + 87 * MB);    // 32 MB, reuses dead K+Vt

  // f32 -> f16 conversions
  cvt_f16<<<4096, 256, 0, stream>>>(x,    x_h,   M_ * E_);
  cvt_f16<<<1024, 256, 0, stream>>>(Wqkv, Wqkvh, 3 * E_ * E_);
  cvt_f16<<<512,  256, 0, stream>>>(Wout, Wouth, E_ * E_);
  cvt_f16<<<1024, 256, 0, stream>>>(W1,   W1h,   4 * E_ * E_);
  cvt_f16<<<1024, 256, 0, stream>>>(W2,   W2h,   4 * E_ * E_);

  // qkv = x @ Wqkv^T + b          (bias, f16 out)
  gemm_wmma<1 | 16><<<dim3(1536 / 128, M_ / 128), 128, 0, stream>>>(
      x_h, Wqkvh, bqkv, nullptr, nullptr, qkv, M_, 3 * E_, E_);

  repack_qkv<<<(M_ * E_) / 256, 256, 0, stream>>>(qkv, Qb, Kb, Vtb);

  // causal attention -> ctx
  attn_wmma<<<dim3(S_ / 16, B_ * H_), 32, 0, stream>>>(Qb, Kb, Vtb, ctx);

  // y1 = x + relu(ctx @ Wout^T + b)   (bias, relu, resid, f32 out)
  gemm_wmma<1 | 2 | 4 | 8><<<dim3(512 / 128, M_ / 128), 128, 0, stream>>>(
      ctx, Wouth, bout, x, y1, nullptr, M_, E_, E_);

  ln_rows<<<M_, 128, 0, stream>>>(y1, g1, be1, x1f, x1h);

  // h1 = relu(x1 @ W1^T + b1)     (bias, relu, f16 out)
  gemm_wmma<1 | 2 | 16><<<dim3(2048 / 128, M_ / 128), 128, 0, stream>>>(
      x1h, W1h, b1, nullptr, nullptr, h1, M_, 4 * E_, E_);

  // y2 = x1 + relu(h1 @ W2^T + b2)
  gemm_wmma<1 | 2 | 4 | 8><<<dim3(512 / 128, M_ / 128), 128, 0, stream>>>(
      h1, W2h, b2, x1f, y2, nullptr, M_, E_, 4 * E_);

  ln_rows<<<M_, 128, 0, stream>>>(y2, g2, be2, out, nullptr);
}
